// SNN_27032524161435
// MI455X (gfx1250) — compile-verified, matrix operand order checked by reading, not run on hardware
//
#include <hip/hip_runtime.h>
#include <hip/hip_bf16.h>
#include <stdint.h>

// ---------------------------------------------------------------------------
// SNN forward for MI455X (gfx1250).
//  - bf16 WMMA (v_wmma_f32_16x16x32_bf16) for all big GEMMs; spikes are {0,1}
//    so bf16 A-operands are exact.
//  - Each LIF layer's 100-step scan runs as ONE persistent kernel:
//      * 32 blocks x 256 threads (8 waves), grid-wide atomic barrier per step
//      * cur/pot membrane state lives in VGPRs across all steps
//      * the block's 64-column slice of R (128KB bf16) is staged into LDS once
//        via global_load_async_to_lds_b128 (+ s_wait_asynccnt), then consumed
//        by ds_load_b128 for every step's WMMA B operand.
// ---------------------------------------------------------------------------

typedef __bf16 bf16_t;
typedef __attribute__((ext_vector_type(16))) __bf16 v16bf;
typedef __attribute__((ext_vector_type(8)))  __bf16 v8bf;
typedef __attribute__((ext_vector_type(8)))  float   v8f;

#define ALPHA_ 0.9048374180359595f   // exp(-0.001/0.010)
#define BETA_  0.8187307530779818f   // exp(-0.001/0.005)

#define B_  256
#define T_  100
#define F_  256
#define H_  1024
#define O_  8

// ---------------------------------------------------------------------------
// 16x32 bf16 operand tile, CDNA5 WMMA A layout (CDNA5 ISA 7.12.2):
// lane<16 -> row, K {0..7,16..23}; lane>=16 -> same row, K {8..15,24..31}.
// Used for A, and for B against pre-transposed [N,K] weights.
// ---------------------------------------------------------------------------
__device__ __forceinline__ v16bf load_tile16(const bf16_t* __restrict__ base,
                                             unsigned ld, int rowBase, int k0,
                                             int lane) {
  const int r  = rowBase + (lane & 15);
  const int kh = (lane >> 4) << 3;  // 0 or 8
  const bf16_t* p = base + (size_t)r * ld + (size_t)(k0 + kh);
  v8bf c0 = *(const v8bf*)(p);
  v8bf c1 = *(const v8bf*)(p + 16);
  v16bf t;
#pragma unroll
  for (int i = 0; i < 8; ++i) { t[i] = c0[i]; t[i + 8] = c1[i]; }
  return t;
}

// ---------------------------------------------------------------------------
// Big GEMM: C[M,N] f32 = A[M,K] bf16 x Bt[N,K] bf16.  Wave -> 16x64 tile.
// Waves within a block share the same B tiles (same nGroup) -> WGP$ hits.
// ---------------------------------------------------------------------------
__global__ void __launch_bounds__(256) wmma_gemm16x64(
    const bf16_t* __restrict__ A, unsigned lda,
    const bf16_t* __restrict__ Bt, unsigned ldb,
    float* __restrict__ C, unsigned ldc, int M, int N, int K) {
  const int lane = threadIdx.x & 31;
  const int wave = (blockIdx.x << 3) + (threadIdx.x >> 5);
  const int mTiles = M >> 4;
  const int nGroup = wave / mTiles;        // 8 waves of a block share nGroup
  const int mTile = wave - nGroup * mTiles;
  if (nGroup >= (N >> 6)) return;          // wave-uniform
  const int mBase = mTile << 4;
  const int nBase = nGroup << 6;

  v8f acc[4] = {v8f{}, v8f{}, v8f{}, v8f{}};
  for (int k0 = 0; k0 < K; k0 += 32) {
    v16bf a = load_tile16(A, lda, mBase, k0, lane);
#pragma unroll
    for (int s = 0; s < 4; ++s) {
      v16bf b = load_tile16(Bt, ldb, nBase + 16 * s, k0, lane);
      acc[s] = __builtin_amdgcn_wmma_f32_16x16x32_bf16(
          false, a, false, b, (short)0, acc[s], false, false);
    }
  }
  const int mRow = mBase + ((lane >> 4) << 3);
  const int nCol = nBase + (lane & 15);
#pragma unroll
  for (int j = 0; j < 8; ++j) {
    float* cr = C + (size_t)(mRow + j) * ldc + nCol;
    cr[0]  = acc[0][j];
    cr[16] = acc[1][j];
    cr[32] = acc[2][j];
    cr[48] = acc[3][j];
  }
}

// ---------------------------------------------------------------------------
// Persistent LIF scan for one layer.  Grid MUST be exactly 32 blocks.
// Block (mb = blk>>4 in {0,1}, nb = blk&15): rows mb*128..+127, cols nb*64..+63.
// Wave w owns 16 rows.  Per step: rec = spk_{t-1} @ R (WMMA, B from LDS),
// then LIF update with cur/pot in registers, then grid barrier.
// ---------------------------------------------------------------------------
__global__ void __launch_bounds__(256) lif_scan(
    const bf16_t* __restrict__ Rt,     // [H,H] bf16, transposed ([n][k])
    const float* __restrict__ hbuf,    // [B,T,H] feed-forward currents
    bf16_t* __restrict__ sbf,          // [B,T,H] bf16 spike history (rw)
    float* __restrict__ s_out,         // [B,T,H] spikes f32 (d_out)
    float* __restrict__ v_out,         // [B,T,H] potentials f32 (d_out)
    unsigned* __restrict__ ctr) {      // grid-barrier counter (zeroed)
  __shared__ __align__(16) bf16_t Bs[64 * H_];  // 128 KB: this block's R slice

  const int lane = threadIdx.x & 31;
  const int wave = threadIdx.x >> 5;
  const int mb = blockIdx.x >> 4;
  const int nb = blockIdx.x & 15;
  const int mBaseW = mb * 128 + wave * 16;   // global row base of this wave
  const int nBaseB = nb * 64;                // global col base of this block

  // ---- stage R slice (contiguous 128KB of Rt) into LDS via async DMA ----
  {
    const char* src = (const char*)(Rt + (size_t)nBaseB * H_);
    const unsigned ldsBase = (unsigned)(uintptr_t)(&Bs[0]);
    const int tid = threadIdx.x;
#pragma unroll
    for (int j = 0; j < 32; ++j) {          // 256 thr * 32 * 16B = 128 KB
      const unsigned off = (unsigned)(tid * 16 + j * 4096);
      const unsigned ldsAddr = ldsBase + off;
      const uint64_t ga = (uint64_t)(src + off);
      asm volatile("global_load_async_to_lds_b128 %0, %1, off"
                   :: "v"(ldsAddr), "v"(ga) : "memory");
    }
    asm volatile("s_wait_asynccnt 0" ::: "memory");
    __syncthreads();
  }

  // B fragment from LDS (cols relative to block slice), WMMA operand layout
  auto loadB = [&](int colBase, int k0) -> v16bf {
    const int c = colBase + (lane & 15);
    const int kh = (lane >> 4) << 3;
    const bf16_t* p = &Bs[(unsigned)c * (unsigned)H_ + (unsigned)(k0 + kh)];
    v8bf c0 = *(const v8bf*)(p);
    v8bf c1 = *(const v8bf*)(p + 16);
    v16bf t;
#pragma unroll
    for (int i = 0; i < 8; ++i) { t[i] = c0[i]; t[i + 8] = c1[i]; }
    return t;
  };

  // membrane state in registers for the whole scan
  float cur[4][8], pot[4][8];
#pragma unroll
  for (int s = 0; s < 4; ++s)
#pragma unroll
    for (int j = 0; j < 8; ++j) { cur[s][j] = 0.f; pot[s][j] = 0.f; }

  const unsigned strideT = (unsigned)(T_ * H_);
  const int rBase = mBaseW + ((lane >> 4) << 3);
  const int cBase = nBaseB + (lane & 15);

  for (int t = 0; t < T_; ++t) {
    v8f acc[4] = {v8f{}, v8f{}, v8f{}, v8f{}};
    if (t > 0) {  // uniform; rec = spk_{t-1} @ R
      const bf16_t* A = sbf + (size_t)(t - 1) * H_;  // row stride T*H
      for (int k0 = 0; k0 < H_; k0 += 32) {
        v16bf a = load_tile16(A, strideT, mBaseW, k0, lane);
#pragma unroll
        for (int s = 0; s < 4; ++s) {
          v16bf b = loadB(16 * s, k0);
          acc[s] = __builtin_amdgcn_wmma_f32_16x16x32_bf16(
              false, a, false, b, (short)0, acc[s], false, false);
        }
      }
    }
    // LIF epilogue (C layout: VGPR j -> M = j + 8*(lane>=16), N = lane&15)
#pragma unroll
    for (int s = 0; s < 4; ++s) {
#pragma unroll
      for (int j = 0; j < 8; ++j) {
        const int b = rBase + j;
        const int n = cBase + (s << 4);
        const size_t oidx = (size_t)b * strideT + (size_t)t * H_ + n;
        const float spk = (pot[s][j] > 1.0f) ? 1.0f : 0.0f;
        const float nc = fmaf(ALPHA_, cur[s][j], hbuf[oidx] + acc[s][j]);
        const float np = fmaf(BETA_, pot[s][j], nc) * (1.0f - spk);
        cur[s][j] = nc;
        pot[s][j] = np;
        s_out[oidx] = spk;
        v_out[oidx] = np;
        sbf[oidx] = (bf16_t)spk;  // exact 0/1
      }
    }
    // grid-wide barrier so every block sees all spikes of step t
    if (t != T_ - 1) {
      __syncthreads();
      if (threadIdx.x == 0) {
        __hip_atomic_fetch_add(ctr, 1u, __ATOMIC_RELEASE,
                               __HIP_MEMORY_SCOPE_AGENT);
        const unsigned tgt = (unsigned)(t + 1) * 32u;
        while (__hip_atomic_load(ctr, __ATOMIC_ACQUIRE,
                                 __HIP_MEMORY_SCOPE_AGENT) < tgt) {
          __builtin_amdgcn_s_sleep(1);
        }
      }
      __syncthreads();
    }
  }
}

// ---------------------------------------------------------------------------
// Small helpers
// ---------------------------------------------------------------------------
__global__ void zero_f32(float* __restrict__ p, size_t n) {
  size_t i = (size_t)blockIdx.x * blockDim.x + threadIdx.x;
  if (i < n) p[i] = 0.0f;
}

__global__ void cvt_bf16(const float* __restrict__ x, bf16_t* __restrict__ y,
                         size_t n) {
  size_t i = (size_t)blockIdx.x * blockDim.x + threadIdx.x;
  if (i < n) y[i] = (bf16_t)x[i];
}

// Wt[n*K + k] = (bf16) W[k*N + n]
__global__ void transpose_cvt(const float* __restrict__ W,
                              bf16_t* __restrict__ Wt, int K, int N) {
  size_t i = (size_t)blockIdx.x * blockDim.x + threadIdx.x;
  if (i >= (size_t)K * N) return;
  int n = (int)(i / K);
  int k = (int)(i - (size_t)n * K);
  Wt[i] = (bf16_t)W[(size_t)k * N + n];
}

// h3[r,o] = sum_h s2bf[r,h] * Wout[h,o]
__global__ void readout_mm(const bf16_t* __restrict__ s2bf,
                           const float* __restrict__ Wout,
                           float* __restrict__ h3) {
  int idx = blockIdx.x * blockDim.x + threadIdx.x;
  if (idx >= B_ * T_ * O_) return;
  const int r = idx >> 3;
  const int o = idx & 7;
  const bf16_t* srow = s2bf + (size_t)r * H_;
  float a0 = 0.f, a1 = 0.f, a2 = 0.f, a3 = 0.f;
  for (int h = 0; h < H_; h += 4) {
    a0 = fmaf((float)srow[h + 0], Wout[(h + 0) * O_ + o], a0);
    a1 = fmaf((float)srow[h + 1], Wout[(h + 1) * O_ + o], a1);
    a2 = fmaf((float)srow[h + 2], Wout[(h + 2) * O_ + o], a2);
    a3 = fmaf((float)srow[h + 3], Wout[(h + 3) * O_ + o], a3);
  }
  h3[idx] = (a0 + a1) + (a2 + a3);
}

__global__ void readout_scan(const float* __restrict__ h3,
                             float* __restrict__ out) {
  int idx = blockIdx.x * blockDim.x + threadIdx.x;
  if (idx >= B_ * O_) return;
  const int b = idx >> 3;
  const int o = idx & 7;
  float c = 0.f, p = 0.f;
  float* obase = out + (size_t)b * (T_ + 1) * O_ + o;
  obase[0] = 0.f;
  for (int t = 0; t < T_; ++t) {
    c = fmaf(ALPHA_, c, h3[((size_t)b * T_ + t) * O_ + o]);
    p = fmaf(BETA_, p, c);
    obase[(size_t)(t + 1) * O_] = p;
  }
}

// ---------------------------------------------------------------------------
// Host launcher
// ---------------------------------------------------------------------------
extern "C" void kernel_launch(void* const* d_in, const int* in_sizes, int n_in,
                              void* d_out, int out_size, void* d_ws,
                              size_t ws_size, hipStream_t stream) {
  (void)in_sizes; (void)n_in; (void)out_size; (void)ws_size;
  const float* inputs = (const float*)d_in[0];  // [B,T,F]
  const float* W0     = (const float*)d_in[1];  // [F,H]
  const float* W1     = (const float*)d_in[2];  // [H,H]
  const float* R0     = (const float*)d_in[3];  // [H,H]
  const float* R1     = (const float*)d_in[4];  // [H,H]
  const float* Wout   = (const float*)d_in[5];  // [H,O]
  float* out = (float*)d_out;

  char* ws = (char*)d_ws;
  size_t off = 0;
  auto take = [&](size_t bytes) -> char* {
    char* p = ws + off;
    off = (off + bytes + 255) & ~(size_t)255;
    return p;
  };
  bf16_t* xbf  = (bf16_t*)take((size_t)B_ * T_ * F_ * 2);
  bf16_t* W0t  = (bf16_t*)take((size_t)H_ * F_ * 2);
  bf16_t* W1t  = (bf16_t*)take((size_t)H_ * H_ * 2);
  bf16_t* R0t  = (bf16_t*)take((size_t)H_ * H_ * 2);
  bf16_t* R1t  = (bf16_t*)take((size_t)H_ * H_ * 2);
  bf16_t* s1bf = (bf16_t*)take((size_t)B_ * T_ * H_ * 2);
  bf16_t* s2bf = (bf16_t*)take((size_t)B_ * T_ * H_ * 2);
  float*  hbuf = (float*)take((size_t)B_ * T_ * H_ * 4);
  float*  h3   = (float*)take((size_t)B_ * T_ * O_ * 4);
  unsigned* ctrs = (unsigned*)take(256);        // ctr0 @ +0, ctr1 @ +64
  unsigned* ctr0 = ctrs;
  unsigned* ctr1 = ctrs + 16;

  float* out0 = out;
  float* s1o = out + (size_t)B_ * (T_ + 1) * O_;
  float* s2o = s1o + (size_t)B_ * T_ * H_;
  float* v1o = s2o + (size_t)B_ * T_ * H_;
  float* v2o = v1o + (size_t)B_ * T_ * H_;

  // 1) zero the grid-barrier counters (ws is poisoned before timing)
  zero_f32<<<dim3(1), dim3(64), 0, stream>>>((float*)ctrs, 64);

  // 2) precision/layout prep
  {
    size_t n = (size_t)B_ * T_ * F_;
    cvt_bf16<<<dim3((unsigned)((n + 255) / 256)), dim3(256), 0, stream>>>(inputs, xbf, n);
  }
  transpose_cvt<<<dim3((F_ * H_ + 255) / 256), dim3(256), 0, stream>>>(W0, W0t, F_, H_);
  transpose_cvt<<<dim3((H_ * H_ + 255) / 256), dim3(256), 0, stream>>>(W1, W1t, H_, H_);
  transpose_cvt<<<dim3((H_ * H_ + 255) / 256), dim3(256), 0, stream>>>(R0, R0t, H_, H_);
  transpose_cvt<<<dim3((H_ * H_ + 255) / 256), dim3(256), 0, stream>>>(R1, R1t, H_, H_);

  // 3) h1 = inputs @ W0
  {
    const int M = B_ * T_, N = H_, K = F_;
    const int waves = (M / 16) * (N / 64);
    wmma_gemm16x64<<<dim3(waves / 8), dim3(256), 0, stream>>>(
        xbf, (unsigned)F_, W0t, (unsigned)F_, hbuf, (unsigned)H_, M, N, K);
  }
  // 4) layer-1 scan: one persistent kernel, 100 steps
  lif_scan<<<dim3(32), dim3(256), 0, stream>>>(R0t, hbuf, s1bf, s1o, v1o, ctr0);

  // 5) h2 = s1 @ W1
  {
    const int M = B_ * T_, N = H_, K = H_;
    const int waves = (M / 16) * (N / 64);
    wmma_gemm16x64<<<dim3(waves / 8), dim3(256), 0, stream>>>(
        s1bf, (unsigned)H_, W1t, (unsigned)H_, hbuf, (unsigned)H_, M, N, K);
  }
  // 6) layer-2 scan
  lif_scan<<<dim3(32), dim3(256), 0, stream>>>(R1t, hbuf, s2bf, s2o, v2o, ctr1);

  // 7) readout
  readout_mm<<<dim3((B_ * T_ * O_ + 255) / 256), dim3(256), 0, stream>>>(s2bf, Wout, h3);
  readout_scan<<<dim3((B_ * O_ + 255) / 256), dim3(256), 0, stream>>>(h3, out0);
}
